// SimpleMaskDecoder_37615323578649
// MI455X (gfx1250) — matrix operand order repaired and checked by reading, not compiled, not run on hardware
//
#include <hip/hip_runtime.h>
#include <cstddef>
#include <cstdint>

// ---------- types ----------
typedef __attribute__((ext_vector_type(16))) _Float16 v16h;
typedef __attribute__((ext_vector_type(8)))  _Float16 v8h;
typedef __attribute__((ext_vector_type(8)))  float    v8f;

union V16 { v16h v; v8h h[2]; };

// ---------- problem constants ----------
#define DDIM   384
#define NHEAD  6
#define DH     64
#define NPAD   2688            // 2563 tokens padded; divisible by 128/64/32
#define MPROWS (2*NPAD)        // 5376
#define NREAL  2563
#define NPATCH 2560

// ---------- GEMM tile config ----------
#define BM 128
#define BN 64
#define BK 32
#define LDS_STRIDE 40          // halves per LDS row (32 + 8 pad, keeps b128 16B aligned)

__device__ __forceinline__ float gelu_f(float x) {
  float x3 = x * x * x;
  return 0.5f * x * (1.0f + tanhf(0.7978845608028654f * (x + 0.044715f * x3)));
}

// Async global->LDS copy of 16 bytes per lane (CDNA5 GLOBAL_LOAD_ASYNC_TO_LDS_B128,
// tracked on ASYNCcnt). On gfx1250 the LDS aperture truncates a generic LDS
// address to its low 32 bits, so the VDST operand is just the truncated pointer.
__device__ __forceinline__ void async_copy_b128(const _Float16* gsrc, _Float16* lds_dst) {
  unsigned ldsa = (unsigned)(uintptr_t)lds_dst;
  asm volatile("global_load_async_to_lds_b128 %0, %1, off"
               :: "v"(ldsa), "v"(gsrc)
               : "memory");
}

__device__ __forceinline__ void wait_async0() {
  asm volatile("s_wait_asynccnt 0x0" ::: "memory");
}

struct GemmArgs {
  const _Float16* A;     // [M][K] row-major f16
  const _Float16* Bt;    // [N][K] row-major f16 (i.e. W transposed)
  const float*    bias;  // [N] or null
  float*          outF;  // f32 out or null
  _Float16*       outH;  // f16 out or null
  const float*    res;   // residual (read at outF layout) or null
  int M, N, K;
  int ldcF, ldcH;
  int ripb, ropb;        // row remap: in-rows-per-batch / out-rows-per-batch (0 = identity)
  int act;               // 0 none, 1 relu, 2 gelu
  int qkv;               // 1 => scatter into q/k/vt buffers
  _Float16 *qb, *kb, *vtb;
  int Npad;
};

__global__ void __launch_bounds__(256) gemm_kernel(const GemmArgs ga) {
  __shared__ _Float16 As[2][BM * LDS_STRIDE];
  __shared__ _Float16 Bs[2][BN * LDS_STRIDE];

  const int tid  = threadIdx.x;
  const int rowBase = blockIdx.y * BM;
  const int colBase = blockIdx.x * BN;
  const int wid  = tid >> 5;
  const int lane = tid & 31;
  const int lr   = lane & 15;      // row (A) / col (B) within tile
  const int lg   = lane >> 4;      // lane group
  const int wm   = (wid >> 1) * 32;
  const int wn   = (wid & 1) * 32;

  // per-thread staging map: A tile = 512 16B chunks (2/thread), B tile = 256 (1/thread)
  const int ar0 = tid >> 2,           ac0 = tid & 3;
  const int ar1 = (tid + 256) >> 2,   ac1 = tid & 3;   // +256 keeps cc identical
  const int br  = tid >> 2,           bc0 = tid & 3;

  auto stage_tile = [&](int k0, int buf) {
    async_copy_b128(ga.A + (size_t)(rowBase + ar0) * ga.K + k0 + ac0 * 8,
                    &As[buf][ar0 * LDS_STRIDE + ac0 * 8]);
    async_copy_b128(ga.A + (size_t)(rowBase + ar1) * ga.K + k0 + ac1 * 8,
                    &As[buf][ar1 * LDS_STRIDE + ac1 * 8]);
    async_copy_b128(ga.Bt + (size_t)(colBase + br) * ga.K + k0 + bc0 * 8,
                    &Bs[buf][br * LDS_STRIDE + bc0 * 8]);
  };

  v8f acc[2][2];
#pragma unroll
  for (int i = 0; i < 2; ++i)
#pragma unroll
    for (int j = 0; j < 2; ++j)
#pragma unroll
      for (int e = 0; e < 8; ++e) acc[i][j][e] = 0.0f;

  const int nk = ga.K / BK;
  stage_tile(0, 0);

  int buf = 0;
  for (int kt = 0; kt < nk; ++kt) {
    wait_async0();          // tile kt resident in LDS[buf] (own wave's copies)
    __syncthreads();        // ...and everyone else's; prior reads of LDS[buf^1] done
    if (kt + 1 < nk) stage_tile((kt + 1) * BK, buf ^ 1);  // overlap with WMMA below

    V16 af[2], bf[2];
#pragma unroll
    for (int t = 0; t < 2; ++t) {
      int ar = wm + t * 16 + lr;
      af[t].h[0] = *(const v8h*)(&As[buf][ar * LDS_STRIDE + lg * 8]);
      af[t].h[1] = *(const v8h*)(&As[buf][ar * LDS_STRIDE + 16 + lg * 8]);
      int bc = wn + t * 16 + lr;
      bf[t].h[0] = *(const v8h*)(&Bs[buf][bc * LDS_STRIDE + lg * 8]);
      bf[t].h[1] = *(const v8h*)(&Bs[buf][bc * LDS_STRIDE + 16 + lg * 8]);
    }
#pragma unroll
    for (int i = 0; i < 2; ++i)
#pragma unroll
      for (int j = 0; j < 2; ++j)
        acc[i][j] = __builtin_amdgcn_wmma_f32_16x16x32_f16(
            false, af[i].v, false, bf[j].v, (short)0, acc[i][j], false, false);
    buf ^= 1;
  }

  // epilogue
#pragma unroll
  for (int i = 0; i < 2; ++i) {
#pragma unroll
    for (int j = 0; j < 2; ++j) {
#pragma unroll
      for (int e = 0; e < 8; ++e) {
        int gm = rowBase + wm + i * 16 + e + lg * 8;
        int gn = colBase + wn + j * 16 + lr;
        float v = acc[i][j][e];
        if (ga.bias) v += ga.bias[gn];
        if (ga.act == 1) v = fmaxf(v, 0.0f);
        else if (ga.act == 2) v = gelu_f(v);
        int om = gm;
        if (ga.ripb) { int b = gm / ga.ripb; om = b * ga.ropb + (gm - b * ga.ripb); }
        if (ga.res) v += ga.res[(size_t)om * ga.ldcF + gn];
        if (ga.qkv) {
          int t  = gn / DDIM;              // 0=q 1=k 2=v
          int h  = (gn % DDIM) >> 6;
          int d  = gn & 63;
          int bb = gm / ga.Npad;
          int n  = gm % ga.Npad;
          size_t hb = (size_t)bb * NHEAD + h;
          _Float16 hv = (_Float16)v;
          if (t == 0)      ga.qb[(hb * ga.Npad + n) * DH + d] = hv;
          else if (t == 1) ga.kb[(hb * ga.Npad + n) * DH + d] = hv;
          else             ga.vtb[(hb * DH + d) * ga.Npad + n] = hv;
        } else {
          if (ga.outF) ga.outF[(size_t)om * ga.ldcF + gn] = v;
          if (ga.outH) ga.outH[(size_t)om * ga.ldcH + gn] = (_Float16)v;
        }
      }
    }
  }
}

// ---------- flash attention: wave = 16 queries x dh 64, key tiles of 32 ----------
__global__ void __launch_bounds__(128) attn_kernel(const _Float16* Q, const _Float16* Kb,
                                                   const _Float16* Vt, _Float16* Out,
                                                   int Npad, int Nreal) {
  __shared__ _Float16 pb[4][16 * LDS_STRIDE];
  const int w    = threadIdx.x >> 5;
  const int lane = threadIdx.x & 31;
  const int lr   = lane & 15;
  const int lg   = lane >> 4;
  const int bh   = blockIdx.y;                 // b*NHEAD + h
  const int qbase = blockIdx.x * 64 + w * 16;

  const _Float16* Qh = Q  + (size_t)bh * Npad * DH;
  const _Float16* Kh = Kb + (size_t)bh * Npad * DH;
  const _Float16* Vh = Vt + (size_t)bh * DH * Npad;

  V16 qf[2];
#pragma unroll
  for (int ks = 0; ks < 2; ++ks) {
    const _Float16* p = Qh + (size_t)(qbase + lr) * DH + ks * 32 + lg * 8;
    qf[ks].h[0] = *(const v8h*)p;
    qf[ks].h[1] = *(const v8h*)(p + 16);
  }

  v8f o[4];
  float m[8], s[8];
#pragma unroll
  for (int t = 0; t < 4; ++t)
#pragma unroll
    for (int e = 0; e < 8; ++e) o[t][e] = 0.0f;
#pragma unroll
  for (int e = 0; e < 8; ++e) { m[e] = -1e30f; s[e] = 0.0f; }

  const int ntiles = Npad / 32;
  for (int kt = 0; kt < ntiles; ++kt) {
    // S = Q Kt tile (16 x 32)
    v8f sa[2];
#pragma unroll
    for (int j = 0; j < 2; ++j)
#pragma unroll
      for (int e = 0; e < 8; ++e) sa[j][e] = 0.0f;
#pragma unroll
    for (int j = 0; j < 2; ++j)
#pragma unroll
      for (int ks = 0; ks < 2; ++ks) {
        V16 bk;
        const _Float16* p = Kh + (size_t)(kt * 32 + j * 16 + lr) * DH + ks * 32 + lg * 8;
        bk.h[0] = *(const v8h*)p;
        bk.h[1] = *(const v8h*)(p + 16);
        sa[j] = __builtin_amdgcn_wmma_f32_16x16x32_f16(
            false, qf[ks].v, false, bk.v, (short)0, sa[j], false, false);
      }
    // scale + key mask
#pragma unroll
    for (int j = 0; j < 2; ++j) {
      int col = kt * 32 + j * 16 + lr;
      float mask = (col < Nreal) ? 0.0f : -1e30f;
#pragma unroll
      for (int e = 0; e < 8; ++e) sa[j][e] = sa[j][e] * 0.125f + mask;
    }
    // online softmax stats (rows live across 16-lane groups)
    float cf[8];
#pragma unroll
    for (int e = 0; e < 8; ++e) {
      float t = fmaxf(sa[0][e], sa[1][e]);
      t = fmaxf(t, __shfl_xor(t, 1, 32));
      t = fmaxf(t, __shfl_xor(t, 2, 32));
      t = fmaxf(t, __shfl_xor(t, 4, 32));
      t = fmaxf(t, __shfl_xor(t, 8, 32));
      float nm = fmaxf(m[e], t);
      cf[e] = __expf(m[e] - nm);
      m[e] = nm;
    }
    float ps[8];
#pragma unroll
    for (int e = 0; e < 8; ++e) ps[e] = 0.0f;
#pragma unroll
    for (int j = 0; j < 2; ++j)
#pragma unroll
      for (int e = 0; e < 8; ++e) {
        float p = __expf(sa[j][e] - m[e]);
        ps[e] += p;
        pb[w][(e + 8 * lg) * LDS_STRIDE + j * 16 + lr] = (_Float16)p;  // C-layout -> LDS
      }
#pragma unroll
    for (int e = 0; e < 8; ++e) {
      float t = ps[e];
      t += __shfl_xor(t, 1, 32);
      t += __shfl_xor(t, 2, 32);
      t += __shfl_xor(t, 4, 32);
      t += __shfl_xor(t, 8, 32);
      s[e] = s[e] * cf[e] + t;
    }
#pragma unroll
    for (int t4 = 0; t4 < 4; ++t4)
#pragma unroll
      for (int e = 0; e < 8; ++e) o[t4][e] *= cf[e];
    // reload P as A-fragment
    V16 pf;
    pf.h[0] = *(const v8h*)(&pb[w][lr * LDS_STRIDE + lg * 8]);
    pf.h[1] = *(const v8h*)(&pb[w][lr * LDS_STRIDE + 16 + lg * 8]);
    // O += P @ V (V stored transposed [dh][Npad])
#pragma unroll
    for (int t4 = 0; t4 < 4; ++t4) {
      V16 bv;
      const _Float16* p = Vh + (size_t)(t4 * 16 + lr) * Npad + kt * 32 + lg * 8;
      bv.h[0] = *(const v8h*)p;
      bv.h[1] = *(const v8h*)(p + 16);
      o[t4] = __builtin_amdgcn_wmma_f32_16x16x32_f16(
          false, pf.v, false, bv.v, (short)0, o[t4], false, false);
    }
  }

  const int bb = bh / NHEAD, h = bh % NHEAD;
#pragma unroll
  for (int t4 = 0; t4 < 4; ++t4)
#pragma unroll
    for (int e = 0; e < 8; ++e) {
      int row = qbase + e + 8 * lg;
      int col = h * DH + t4 * 16 + lr;
      Out[((size_t)bb * Npad + row) * DDIM + col] = (_Float16)(o[t4][e] / s[e]);
    }
}

// ---------- small helper kernels ----------
__global__ void transpose_f32_to_h(const float* __restrict__ src, _Float16* __restrict__ dst,
                                   int R, int C) {
  int i = blockIdx.x * 256 + threadIdx.x;
  if (i >= R * C) return;
  int r = i / C, c = i % C;
  dst[(size_t)c * R + r] = (_Float16)src[i];
}

__global__ void cvt_f32_to_h(const float* __restrict__ src, _Float16* __restrict__ dst, int n) {
  int i = blockIdx.x * 256 + threadIdx.x;
  if (i < n) dst[i] = (_Float16)src[i];
}

__global__ void pos1_kernel(const int* __restrict__ pc, const float* __restrict__ w1,
                            const float* __restrict__ b1, _Float16* __restrict__ ph) {
  int i = blockIdx.x * 256 + threadIdx.x;
  if (i >= NPATCH * DDIM) return;
  int n = i / DDIM, d = i % DDIM;
  float v = (float)pc[n * 2] * w1[d] + (float)pc[n * 2 + 1] * w1[DDIM + d] + b1[d];
  ph[i] = (_Float16)fmaxf(v, 0.0f);
}

__global__ void add_pe_cls(float* __restrict__ X, const float* __restrict__ pe,
                           const float* __restrict__ cls) {
  int i = blockIdx.x * 256 + threadIdx.x;
  if (i >= 2 * NPAD * DDIM) return;
  int col = i % DDIM;
  int n = (i / DDIM) % NPAD;
  if (n < NPATCH)          X[i] += pe[(size_t)n * DDIM + col];
  else if (n < NREAL)      X[i] = cls[(n - NPATCH) * DDIM + col];
  else                     X[i] = 0.0f;
}

// rowwise layernorm over D=384, one wave per row
__global__ void __launch_bounds__(256) ln_kernel(const float* __restrict__ X,
                                                 const float* __restrict__ g,
                                                 const float* __restrict__ b,
                                                 _Float16* __restrict__ o16,
                                                 float* __restrict__ o32, int rows) {
  int w = threadIdx.x >> 5, lane = threadIdx.x & 31;
  int row = blockIdx.x * 8 + w;
  if (row >= rows) return;
  const float* x = X + (size_t)row * DDIM;
  float v[12], sum = 0.0f, sq = 0.0f;
#pragma unroll
  for (int i = 0; i < 12; ++i) {
    v[i] = x[lane + i * 32];
    sum += v[i];
    sq += v[i] * v[i];
  }
#pragma unroll
  for (int off = 1; off < 32; off <<= 1) {
    sum += __shfl_xor(sum, off, 32);
    sq  += __shfl_xor(sq, off, 32);
  }
  float mean = sum * (1.0f / DDIM);
  float var  = sq * (1.0f / DDIM) - mean * mean;
  float rs   = rsqrtf(var + 1e-5f);
#pragma unroll
  for (int i = 0; i < 12; ++i) {
    int c = lane + i * 32;
    float y = (v[i] - mean) * rs * g[c] + b[c];
    o16[(size_t)row * DDIM + c] = (_Float16)y;
    if (o32) o32[(size_t)row * DDIM + c] = y;
  }
}

// cls features: 6 rows of (xn @ proj_classes), L2-normalized
__global__ void __launch_bounds__(384) cls_kernel(const float* __restrict__ xnf,
                                                  const float* __restrict__ Wc,
                                                  float* __restrict__ clsn) {
  __shared__ float red[384];
  int bc = blockIdx.x;           // b*3 + c
  int bb = bc / 3, c = bc % 3;
  int j = threadIdx.x;
  const float* xr = xnf + ((size_t)bb * NPAD + NPATCH + c) * DDIM;
  float acc = 0.0f;
  for (int k = 0; k < DDIM; ++k) acc += xr[k] * Wc[k * DDIM + j];
  red[j] = acc * acc;
  __syncthreads();
  if (j < 128) red[j] += red[j + 256];
  __syncthreads();
  for (int off = 128; off > 0; off >>= 1) {
    if (j < off) red[j] += red[j + off];
    __syncthreads();
  }
  float rn = rsqrtf(red[0] + 1e-30f);
  clsn[(size_t)bc * DDIM + j] = acc * rn;
}

// cosine masks + 3-wide layernorm; one wave per (b, patch)
__global__ void __launch_bounds__(256) masks_kernel(const float* __restrict__ pbuf,
                                                    const float* __restrict__ clsn,
                                                    const float* __restrict__ mg,
                                                    const float* __restrict__ mb,
                                                    float* __restrict__ masks) {
  int w = threadIdx.x >> 5, lane = threadIdx.x & 31;
  int idx = blockIdx.x * 8 + w;
  if (idx >= 2 * NPATCH) return;
  int bb = idx / NPATCH, n = idx % NPATCH;
  const float* pr = pbuf + ((size_t)bb * NPAD + n) * DDIM;
  const float* c0 = clsn + (size_t)(bb * 3) * DDIM;
  const float* c1 = c0 + DDIM;
  const float* c2 = c1 + DDIM;
  float d0 = 0, d1 = 0, d2 = 0, ss = 0;
#pragma unroll
  for (int i = 0; i < 12; ++i) {
    int c = lane + i * 32;
    float v = pr[c];
    ss += v * v;
    d0 += v * c0[c];
    d1 += v * c1[c];
    d2 += v * c2[c];
  }
#pragma unroll
  for (int off = 1; off < 32; off <<= 1) {
    ss += __shfl_xor(ss, off, 32);
    d0 += __shfl_xor(d0, off, 32);
    d1 += __shfl_xor(d1, off, 32);
    d2 += __shfl_xor(d2, off, 32);
  }
  if (lane == 0) {
    float rn = rsqrtf(ss + 1e-30f);
    float e0 = d0 * rn, e1 = d1 * rn, e2 = d2 * rn;
    float mean = (e0 + e1 + e2) * (1.0f / 3.0f);
    float q0 = e0 - mean, q1 = e1 - mean, q2 = e2 - mean;
    float var = (q0 * q0 + q1 * q1 + q2 * q2) * (1.0f / 3.0f);
    float rs = rsqrtf(var + 1e-5f);
    float* o = masks + (size_t)idx * 3;
    o[0] = q0 * rs * mg[0] + mb[0];
    o[1] = q1 * rs * mg[1] + mb[1];
    o[2] = q2 * rs * mg[2] + mb[2];
  }
}

// scatter (expressed as a gather: each 64x64 cell written by exactly one patch)
__global__ void gather_kernel(const float* __restrict__ masks, float* __restrict__ img) {
  int i = blockIdx.x * 256 + threadIdx.x;  // 2*3*64*64
  if (i >= 2 * 3 * 64 * 64) return;
  int C = i & 63, R = (i >> 6) & 63;
  int c = (i >> 12) % 3, bb = i / (3 * 4096);
  int n;
  if (R < 32) {
    n = (R >> 1) * 32 + (C >> 1);                       // scale-0 patch, 2x2 footprint
  } else {
    int dr = R & 1, dc = C & 1, r = R >> 1;             // scale-1 child cell
    n = 512 + (dr * 2 + dc) * 512 + (r - 16) * 32 + (C >> 1);
  }
  img[i] = masks[((size_t)bb * NPATCH + n) * 3 + c];
}

// bilinear 64 -> 512 (half-pixel centers, edge clamp)
__global__ void resize_kernel(const float* __restrict__ img, float* __restrict__ out) {
  int i = blockIdx.x * 256 + threadIdx.x;  // 2*3*512*512
  if (i >= 2 * 3 * 512 * 512) return;
  int x = i & 511, y = (i >> 9) & 511, pl = i >> 18;
  float fy = (y + 0.5f) * 0.125f - 0.5f;
  float fx = (x + 0.5f) * 0.125f - 0.5f;
  int y0 = (int)floorf(fy), x0 = (int)floorf(fx);
  float wy = fy - (float)y0, wx = fx - (float)x0;
  int y1 = y0 + 1, x1 = x0 + 1;
  y0 = y0 < 0 ? 0 : (y0 > 63 ? 63 : y0);
  y1 = y1 < 0 ? 0 : (y1 > 63 ? 63 : y1);
  x0 = x0 < 0 ? 0 : (x0 > 63 ? 63 : x0);
  x1 = x1 < 0 ? 0 : (x1 > 63 ? 63 : x1);
  const float* p = img + (size_t)pl * 4096;
  float v = (1.0f - wy) * ((1.0f - wx) * p[y0 * 64 + x0] + wx * p[y0 * 64 + x1]) +
            wy * ((1.0f - wx) * p[y1 * 64 + x0] + wx * p[y1 * 64 + x1]);
  out[i] = v;
}

// ---------- host orchestration ----------
extern "C" void kernel_launch(void* const* d_in, const int* in_sizes, int n_in,
                              void* d_out, int out_size, void* d_ws, size_t ws_size,
                              hipStream_t stream) {
  const float* x      = (const float*)d_in[0];
  const int*   pc     = (const int*)d_in[1];
  const float* pdw    = (const float*)d_in[2];   // proj_dec_w (768,384)
  const float* pdb    = (const float*)d_in[3];
  const float* pw1    = (const float*)d_in[4];
  const float* pb1    = (const float*)d_in[5];
  const float* pw2    = (const float*)d_in[6];
  const float* pb2    = (const float*)d_in[7];
  const float* clsemb = (const float*)d_in[8];
  const float* ln1g   = (const float*)d_in[9];
  const float* ln1b   = (const float*)d_in[10];
  const float* qkvw   = (const float*)d_in[11];  // (2,384,1152)
  const float* qkvb   = (const float*)d_in[12];
  const float* apw    = (const float*)d_in[13];  // (2,384,384)
  const float* apb    = (const float*)d_in[14];
  const float* ln2g   = (const float*)d_in[15];
  const float* ln2b   = (const float*)d_in[16];
  const float* f1w    = (const float*)d_in[17];  // (2,384,1536)
  const float* f1b    = (const float*)d_in[18];
  const float* f2w    = (const float*)d_in[19];  // (2,1536,384)
  const float* f2b    = (const float*)d_in[20];
  const float* dng    = (const float*)d_in[21];
  const float* dnb    = (const float*)d_in[22];
  const float* ppw    = (const float*)d_in[23];  // proj_patch
  const float* pcw    = (const float*)d_in[24];  // proj_classes
  const float* mng    = (const float*)d_in[25];
  const float* mnb    = (const float*)d_in[26];
  float* out = (float*)d_out;

  // workspace carve-up
  char* wsp = (char*)d_ws;
  size_t off = 0;
  auto alloc = [&](size_t bytes) -> char* {
    char* p = wsp + off;
    off += (bytes + 255) & ~(size_t)255;
    return p;
  };
  _Float16* x16    = (_Float16*)alloc((size_t)2 * NPATCH * 768 * 2);
  _Float16* wdecT  = (_Float16*)alloc((size_t)384 * 768 * 2);
  _Float16* pw2T   = (_Float16*)alloc((size_t)384 * 384 * 2);
  _Float16* qkvT   = (_Float16*)alloc((size_t)2 * 1152 * 384 * 2);
  _Float16* apT    = (_Float16*)alloc((size_t)2 * 384 * 384 * 2);
  _Float16* f1T    = (_Float16*)alloc((size_t)2 * 1536 * 384 * 2);
  _Float16* f2T    = (_Float16*)alloc((size_t)2 * 384 * 1536 * 2);
  _Float16* ppT    = (_Float16*)alloc((size_t)384 * 384 * 2);
  _Float16* ph16   = (_Float16*)alloc((size_t)NPATCH * 384 * 2);
  float*    pe     = (float*)alloc((size_t)NPATCH * 384 * 4);
  float*    X      = (float*)alloc((size_t)MPROWS * 384 * 4);
  _Float16* h16    = (_Float16*)alloc((size_t)MPROWS * 384 * 2);
  _Float16* Qb     = (_Float16*)alloc((size_t)2 * NHEAD * NPAD * DH * 2);
  _Float16* Kbf    = (_Float16*)alloc((size_t)2 * NHEAD * NPAD * DH * 2);
  _Float16* Vtb    = (_Float16*)alloc((size_t)2 * NHEAD * DH * NPAD * 2);
  _Float16* aout   = (_Float16*)alloc((size_t)MPROWS * 384 * 2);
  _Float16* hmid   = (_Float16*)alloc((size_t)MPROWS * 1536 * 2);
  _Float16* xn16   = (_Float16*)alloc((size_t)MPROWS * 384 * 2);
  float*    xnf    = (float*)alloc((size_t)MPROWS * 384 * 4);
  float*    pbuf   = (float*)alloc((size_t)MPROWS * 384 * 4);
  float*    clsn   = (float*)alloc((size_t)6 * 384 * 4);
  float*    masks  = (float*)alloc((size_t)2 * NPATCH * 3 * 4);
  float*    img64  = (float*)alloc((size_t)2 * 3 * 64 * 64 * 4);

  auto tp = [&](const float* s, _Float16* d, int R, int C) {
    int n = R * C;
    transpose_f32_to_h<<<(n + 255) / 256, 256, 0, stream>>>(s, d, R, C);
  };
  auto gemm = [&](const _Float16* A, const _Float16* Bt, const float* bias,
                  float* oF, _Float16* oH, const float* res,
                  int M, int N, int K, int ldcF, int ldcH,
                  int ripb, int ropb, int act, int qkv) {
    GemmArgs ga;
    ga.A = A; ga.Bt = Bt; ga.bias = bias; ga.outF = oF; ga.outH = oH; ga.res = res;
    ga.M = M; ga.N = N; ga.K = K; ga.ldcF = ldcF; ga.ldcH = ldcH;
    ga.ripb = ripb; ga.ropb = ropb; ga.act = act; ga.qkv = qkv;
    ga.qb = Qb; ga.kb = Kbf; ga.vtb = Vtb; ga.Npad = NPAD;
    gemm_kernel<<<dim3(N / BN, M / BM), 256, 0, stream>>>(ga);
  };

  // --- pack weights (f32 -> f16, transposed [N][K]) + convert x
  tp(pdw, wdecT, 768, 384);
  tp(pw2, pw2T, 384, 384);
  tp(qkvw,              qkvT,               384, 1152);
  tp(qkvw + 384 * 1152, qkvT + 1152 * 384,  384, 1152);
  tp(apw,             apT,             384, 384);
  tp(apw + 384 * 384, apT + 384 * 384, 384, 384);
  tp(f1w,              f1T,              384, 1536);
  tp(f1w + 384 * 1536, f1T + 1536 * 384, 384, 1536);
  tp(f2w,              f2T,              1536, 384);
  tp(f2w + 1536 * 384, f2T + 384 * 1536, 1536, 384);
  tp(ppw, ppT, 384, 384);
  {
    int n = 2 * NPATCH * 768;
    cvt_f32_to_h<<<(n + 255) / 256, 256, 0, stream>>>(x, x16, n);
  }
  pos1_kernel<<<(NPATCH * DDIM + 255) / 256, 256, 0, stream>>>(pc, pw1, pb1, ph16);

  // --- embed: X[b, n<2560] = x @ Wdec + b   (row remap 2560 -> 2688)
  gemm(x16, wdecT, pdb, X, nullptr, nullptr, 2 * NPATCH, 384, 768, 384, 384,
       NPATCH, NPAD, 0, 0);
  // pe = relu(pc@w1+b1) @ w2 + b2
  gemm(ph16, pw2T, pb2, pe, nullptr, nullptr, NPATCH, 384, 384, 384, 384, 0, 0, 0, 0);
  add_pe_cls<<<(2 * NPAD * DDIM + 255) / 256, 256, 0, stream>>>(X, pe, clsemb);

  // --- transformer layers
  for (int l = 0; l < 2; ++l) {
    ln_kernel<<<MPROWS / 8, 256, 0, stream>>>(X, ln1g + l * 384, ln1b + l * 384,
                                              h16, nullptr, MPROWS);
    gemm(h16, qkvT + (size_t)l * 1152 * 384, qkvb + l * 1152, nullptr, nullptr, nullptr,
         MPROWS, 1152, 384, 0, 0, 0, 0, 0, /*qkv=*/1);
    attn_kernel<<<dim3(NPAD / 64, 2 * NHEAD), 128, 0, stream>>>(Qb, Kbf, Vtb, aout,
                                                                NPAD, NREAL);
    gemm(aout, apT + (size_t)l * 384 * 384, apb + l * 384, X, nullptr, /*res=*/X,
         MPROWS, 384, 384, 384, 384, 0, 0, 0, 0);
    ln_kernel<<<MPROWS / 8, 256, 0, stream>>>(X, ln2g + l * 384, ln2b + l * 384,
                                              h16, nullptr, MPROWS);
    gemm(h16, f1T + (size_t)l * 1536 * 384, f1b + l * 1536, nullptr, hmid, nullptr,
         MPROWS, 1536, 384, 1536, 1536, 0, 0, /*gelu=*/2, 0);
    gemm(hmid, f2T + (size_t)l * 384 * 1536, f2b + l * 384, X, nullptr, /*res=*/X,
         MPROWS, 384, 1536, 384, 384, 0, 0, 0, 0);
  }

  // --- decode head
  ln_kernel<<<MPROWS / 8, 256, 0, stream>>>(X, dng, dnb, xn16, xnf, MPROWS);
  gemm(xn16, ppT, nullptr, pbuf, nullptr, nullptr, MPROWS, 384, 384, 384, 384, 0, 0, 0, 0);
  cls_kernel<<<6, 384, 0, stream>>>(xnf, pcw, clsn);
  masks_kernel<<<(2 * NPATCH) / 8, 256, 0, stream>>>(pbuf, clsn, mng, mnb, masks);
  gather_kernel<<<(2 * 3 * 64 * 64 + 255) / 256, 256, 0, stream>>>(masks, img64);
  resize_kernel<<<(2 * 3 * 512 * 512 + 255) / 256, 256, 0, stream>>>(img64, out);

  (void)in_sizes; (void)n_in; (void)out_size; (void)ws_size;
}
